// Up_block_83631603187757
// MI455X (gfx1250) — matrix operand order repaired and checked by reading, not compiled, not run on hardware
//
#include <hip/hip_runtime.h>
#include <hip/hip_bf16.h>

typedef __bf16 bf16_t;
typedef __attribute__((ext_vector_type(16))) __bf16 v16bf;
typedef __attribute__((ext_vector_type(8)))  float  v8f;

#define B_   2
#define C_   256
#define N_   4096   // 64*64
#define N1_  1024   // 32*32
#define KNN  16
#define EPSB 1e-5f

union BFV { v16bf v; uint4 q[2]; };
union PK8 { uint4 q; bf16_t h[8]; };

// ---- WMMA operand fragment loads (bf16, 16x16x32) -------------------------
// A (16xK slice): lane L -> row mbase+(L&15); low lanes K{k..k+7, k+16..k+23},
// high lanes K{k+8..k+15, k+24..k+31}.
__device__ __forceinline__ v16bf load_a_frag(const bf16_t* __restrict__ A, int lda,
                                             int mbase, int kbase, int lane) {
  const int row = mbase + (lane & 15);
  const int hi  = (lane >> 4) & 1;
  const bf16_t* p = A + (size_t)row * lda + kbase + hi * 8;
  BFV r;
  r.q[0] = *reinterpret_cast<const uint4*>(p);
  r.q[1] = *reinterpret_cast<const uint4*>(p + 16);
  return r.v;
}
// B (Kx16 slice) from an (n, K)-row-major matrix: lane L -> col nbase+(L&15);
// low lanes K{k..k+7, k+16..k+23}, high lanes K{k+8..k+15, k+24..k+31}.
__device__ __forceinline__ v16bf load_b_frag(const bf16_t* __restrict__ Bm, int ldb,
                                             int nbase, int kbase, int lane) {
  const int col = nbase + (lane & 15);
  const int hi  = (lane >> 4) & 1;
  const bf16_t* p = Bm + (size_t)col * ldb + kbase + hi * 8;
  BFV r;
  r.q[0] = *reinterpret_cast<const uint4*>(p);
  r.q[1] = *reinterpret_cast<const uint4*>(p + 16);
  return r.v;
}

template<int KTOT>
__device__ __forceinline__ v8f gemm_tile(const bf16_t* __restrict__ A, int lda, int mbase,
                                         const bf16_t* __restrict__ Bm, int ldb, int nbase,
                                         int lane) {
  v8f acc = {};
#pragma unroll
  for (int k = 0; k < KTOT; k += 32) {
    v16bf a = load_a_frag(A, lda, mbase, k, lane);
    v16bf b = load_b_frag(Bm, ldb, nbase, k, lane);
    acc = __builtin_amdgcn_wmma_f32_16x16x32_bf16(false, a, false, b, (short)0, acc,
                                                  false, false);
  }
  return acc;
}

// ---- prep: transpose inputs to channel-last bf16 --------------------------
__global__ void k_pack_x(const float* __restrict__ x1, const float* __restrict__ x2,
                         bf16_t* __restrict__ X1t, bf16_t* __restrict__ U) {
  int i = blockIdx.x * blockDim.x + threadIdx.x;
  const int tot1 = B_ * C_ * N1_;
  const int tot2 = B_ * C_ * N_;
  if (i < tot1) {                      // x1 (b,c,p) -> X1t (b,p,c)
    int p = i & (N1_ - 1); int c = (i >> 10) & 255; int b = i >> 18;
    X1t[((size_t)b * N1_ + p) * C_ + c] = (bf16_t)x1[i];
  } else {
    int j = i - tot1;
    if (j < tot2) {                    // x2 (b,c,p) -> U (b,p,c)  (low half)
      int p = j & (N_ - 1); int c = (j >> 12) & 255; int b = j >> 20;
      U[((size_t)b * N_ + p) * 512 + c] = (bf16_t)x2[j];
    }
  }
}

__global__ void k_pack_w(const float* __restrict__ up_tw, const float* __restrict__ upc_w,
                         const float* __restrict__ ga_w,
                         bf16_t* __restrict__ wt4, bf16_t* __restrict__ wab,
                         bf16_t* __restrict__ gaw) {
  int i = blockIdx.x * blockDim.x + threadIdx.x;
  if (i < 4 * C_ * C_) {               // wt4[g][o][i] = up_tw[i][o][dh][dw]
    int g = i >> 16; int r = i & 65535; int o = r >> 8; int ii = r & 255;
    wt4[i] = (bf16_t)up_tw[((ii * C_ + o) * 2 + (g >> 1)) * 2 + (g & 1)];
  } else {
    int j = i - 4 * C_ * C_;
    if (j < C_ * 512)       wab[j] = (bf16_t)upc_w[j];
    else if (j < 2 * C_ * 512) gaw[j - C_ * 512] = (bf16_t)ga_w[j - C_ * 512];
  }
}

// fold attention vector through the (linear) embedding
__global__ void k_attfold(const float* __restrict__ emb_w, const float* __restrict__ emb_b,
                          const float* __restrict__ att_w,
                          float* __restrict__ va1, float* __restrict__ va2,
                          float* __restrict__ c12) {
  int c = threadIdx.x;
  float s1 = 0.f, s2 = 0.f;
  for (int o = 0; o < C_; ++o) {
    float e = emb_w[o * C_ + c];
    s1 += att_w[o] * e;
    s2 += att_w[C_ + o] * e;
  }
  va1[c] = s1; va2[c] = s2;
  if (c == 0) {
    float t1 = 0.f, t2 = 0.f;
    for (int o = 0; o < C_; ++o) { t1 += att_w[o] * emb_b[o]; t2 += att_w[C_ + o] * emb_b[o]; }
    c12[0] = t1; c12[1] = t2;
  }
}

// ---- ConvTranspose2d(2,2): 4 parity GEMMs, scatter bf16 into concat -------
__global__ void k_up(const bf16_t* __restrict__ X1t, const bf16_t* __restrict__ wt4,
                     const float* __restrict__ up_tb, bf16_t* __restrict__ U) {
  int wid = (blockIdx.x * blockDim.x + threadIdx.x) >> 5;
  int lane = threadIdx.x & 31;
  int ot = wid & 15; wid >>= 4;
  int pt = wid & 63; wid >>= 6;
  int g  = wid & 3;
  int b  = wid >> 2;
  const bf16_t* A  = X1t + (size_t)b * N1_ * C_;
  const bf16_t* Bm = wt4 + (size_t)g * C_ * C_;
  v8f acc = gemm_tile<C_>(A, C_, pt * 16, Bm, C_, ot * 16, lane);
  int o  = ot * 16 + (lane & 15);
  int hi = (lane >> 4) & 1;
  float bias = up_tb[o];
  int dh = g >> 1, dw = g & 1;
#pragma unroll
  for (int v = 0; v < 8; ++v) {
    int p2 = pt * 16 + v + hi * 8;
    int h2 = p2 >> 5, w2 = p2 & 31;
    int pout = (2 * h2 + dh) * 64 + (2 * w2 + dw);
    U[((size_t)b * N_ + pout) * 512 + 256 + o] = (bf16_t)(acc[v] + bias);
  }
}

// ---- generic K=512 GEMM + bias, channel-last f32 out (conv1 & ga conv) ----
__global__ void k_gemm_bias(const bf16_t* __restrict__ Aall, const bf16_t* __restrict__ W,
                            const float* __restrict__ bias, float* __restrict__ Out) {
  int wid = (blockIdx.x * blockDim.x + threadIdx.x) >> 5;
  int lane = threadIdx.x & 31;
  int ot = wid & 15;  wid >>= 4;
  int pt = wid & 255; wid >>= 8;
  int b  = wid;
  const bf16_t* A = Aall + (size_t)b * N_ * 512;
  v8f acc = gemm_tile<512>(A, 512, pt * 16, W, 512, ot * 16, lane);
  int o  = ot * 16 + (lane & 15);
  int hi = (lane >> 4) & 1;
  float bo = bias[o];
#pragma unroll
  for (int v = 0; v < 8; ++v) {
    int p = pt * 16 + v + hi * 8;
    Out[((size_t)b * N_ + p) * C_ + o] = acc[v] + bo;
  }
}

// ---- deterministic per-channel BN stats over (b,p) ------------------------
__global__ void k_colstats(const float* __restrict__ Zc, float* __restrict__ mean,
                           float* __restrict__ rstd) {
  __shared__ float ss[256], sq2[256];
  int o = blockIdx.x, t = threadIdx.x;
  float s = 0.f, q = 0.f;
  for (int i = t; i < B_ * N_; i += 256) {
    float z = Zc[(size_t)i * C_ + o];
    s += z; q += z * z;
  }
  ss[t] = s; sq2[t] = q; __syncthreads();
  for (int st = 128; st > 0; st >>= 1) {
    if (t < st) { ss[t] += ss[t + st]; sq2[t] += sq2[t + st]; }
    __syncthreads();
  }
  if (t == 0) {
    float m = ss[0] / (float)(B_ * N_);
    float v = sq2[0] / (float)(B_ * N_) - m * m;
    mean[o] = m; rstd[o] = rsqrtf(v + EPSB);
  }
}

__global__ void k_bnrelu(const float* __restrict__ Z, const float* __restrict__ mean,
                         const float* __restrict__ rstd, const float* __restrict__ g,
                         const float* __restrict__ bet, float* __restrict__ Xf,
                         bf16_t* __restrict__ U2) {
  int i = blockIdx.x * blockDim.x + threadIdx.x;
  if (i >= B_ * N_ * C_) return;
  int o = i & 255; int row = i >> 8;
  float x = (Z[i] - mean[o]) * rstd[o] * g[o] + bet[o];
  x = fmaxf(x, 0.f);
  Xf[i] = x;
  U2[(size_t)row * 512 + o] = (bf16_t)x;   // xn half of final concat
}

// per-(b,channel) L2 norm over points -> 1/max(norm,1e-12)
__global__ void k_colnorm(const float* __restrict__ Xf, float* __restrict__ inv) {
  __shared__ float ss[256];
  int bo = blockIdx.x; int b = bo >> 8, o = bo & 255; int t = threadIdx.x;
  float q = 0.f;
  for (int p = t; p < N_; p += 256) {
    float x = Xf[((size_t)b * N_ + p) * C_ + o];
    q += x * x;
  }
  ss[t] = q; __syncthreads();
  for (int st = 128; st > 0; st >>= 1) { if (t < st) ss[t] += ss[t + st]; __syncthreads(); }
  if (t == 0) inv[bo] = 1.f / fmaxf(sqrtf(ss[0]), 1e-12f);
}

// normalized rows (bf16), sq[n], folded attention scalars a1/a2
__global__ void k_rowstats(const float* __restrict__ Xf, const float* __restrict__ inv,
                           const float* __restrict__ va1, const float* __restrict__ va2,
                           const float* __restrict__ c12,
                           bf16_t* __restrict__ NA, float* __restrict__ sq,
                           float* __restrict__ a1, float* __restrict__ a2) {
  int wid = (blockIdx.x * blockDim.x + threadIdx.x) >> 5;  // row = b*N+p
  int lane = threadIdx.x & 31;
  int b = wid >> 12;
  const float* xrow = Xf + (size_t)wid * C_;
  const float* ivb  = inv + b * C_;
  float ssq = 0.f, s1 = 0.f, s2 = 0.f;
  PK8 pk;
#pragma unroll
  for (int j = 0; j < 8; ++j) {
    int c = lane * 8 + j;
    float x = xrow[c];
    float na = x * ivb[c];
    pk.h[j] = (bf16_t)na;
    ssq += na * na; s1 += va1[c] * x; s2 += va2[c] * x;
  }
  *reinterpret_cast<uint4*>(NA + (size_t)wid * C_ + lane * 8) = pk.q;
#pragma unroll
  for (int m = 16; m >= 1; m >>= 1) {
    ssq += __shfl_xor(ssq, m, 32);
    s1  += __shfl_xor(s1,  m, 32);
    s2  += __shfl_xor(s2,  m, 32);
  }
  if (lane == 0) { sq[wid] = ssq; a1[wid] = s1 + c12[0]; a2[wid] = s2 + c12[1]; }
}

// ---- fused Gram (WMMA) + streaming top-16 ---------------------------------
__global__ __launch_bounds__(128)
void k_knn(const bf16_t* __restrict__ NA, const float* __restrict__ sq,
           int* __restrict__ nn) {
  __shared__ float keybuf[16][512];
  int wave = threadIdx.x >> 5;
  int lane = threadIdx.x & 31;
  int b = blockIdx.x >> 8;
  int nbase = (blockIdx.x & 255) * 16;
  const bf16_t* NAb = NA + (size_t)b * N_ * C_;
  const float*  sqb = sq + b * N_;
  v16bf afr[8];
#pragma unroll
  for (int k = 0; k < 8; ++k) afr[k] = load_a_frag(NAb, C_, nbase, k * 32, lane);
  int hi = (lane >> 4) & 1, lcol = lane & 15;
  float kv[KNN]; int ki[KNN];
#pragma unroll
  for (int j = 0; j < KNN; ++j) { kv[j] = -3.4e38f; ki[j] = 0; }
  for (int chunk = 0; chunk < 8; ++chunk) {
    for (int t = 0; t < 8; ++t) {
      int mtile = chunk * 32 + t * 4 + wave;
      int mbase = mtile * 16;
      __builtin_prefetch(NAb + (size_t)((mbase + 64) & (N_ - 1)) * C_, 0, 1);
      v8f acc = {};
#pragma unroll
      for (int k = 0; k < 8; ++k) {
        v16bf bf = load_b_frag(NAb, C_, mbase, k * 32, lane);
        acc = __builtin_amdgcn_wmma_f32_16x16x32_bf16(false, afr[k], false, bf, (short)0,
                                                      acc, false, false);
      }
      int mcol = mbase + lcol;
      float sqm = sqb[mcol];
      int mloc = mbase - chunk * 512 + lcol;
#pragma unroll
      for (int v = 0; v < 8; ++v)
        keybuf[v + hi * 8][mloc] = 2.f * acc[v] - sqm;   // max key == min dist
    }
    __syncthreads();
    if (threadIdx.x < 16) {
      int row = threadIdx.x;
      for (int j = 0; j < 512; ++j) {
        float kkey = keybuf[row][j];
        if (kkey > kv[KNN - 1]) {       // strict > : earlier index wins ties
          kv[KNN - 1] = kkey; ki[KNN - 1] = chunk * 512 + j;
#pragma unroll
          for (int s = KNN - 1; s > 0; --s) {
            if (kv[s] > kv[s - 1]) {
              float tv = kv[s]; kv[s] = kv[s - 1]; kv[s - 1] = tv;
              int ti = ki[s];  ki[s] = ki[s - 1];  ki[s - 1] = ti;
            }
          }
        }
      }
    }
    __syncthreads();
  }
  if (threadIdx.x < 16) {
    int row = threadIdx.x;
#pragma unroll
    for (int j = 0; j < KNN; ++j)
      nn[((size_t)b * N_ + nbase + row) * KNN + j] = ki[j];
  }
}

// ---- attention softmax over K=16 + weighted aggregation -------------------
__global__ void k_attn_agg(const float* __restrict__ Xf, const int* __restrict__ nn,
                           const float* __restrict__ a1, const float* __restrict__ a2,
                           const float* __restrict__ att_b, bf16_t* __restrict__ U2) {
  int wid = (blockIdx.x * blockDim.x + threadIdx.x) >> 5;  // row = b*N+p
  int lane = threadIdx.x & 31;
  int b = wid >> 12;
  int m = nn[(size_t)wid * KNN + (lane & 15)];
  float l = a1[wid] + a2[b * N_ + m] + att_b[0];
  l = (l > 0.f) ? l : 0.1f * l;                            // leaky_relu(0.1)
  float mx = l;
#pragma unroll
  for (int s = 8; s >= 1; s >>= 1) mx = fmaxf(mx, __shfl_xor(mx, s, 32));
  float e = __expf(l - mx);
  float se = e;
#pragma unroll
  for (int s = 8; s >= 1; s >>= 1) se += __shfl_xor(se, s, 32);
  float w = e / se;
  float acc[8];
#pragma unroll
  for (int j = 0; j < 8; ++j) acc[j] = 0.f;
#pragma unroll
  for (int k = 0; k < KNN; ++k) {
    float wk = __shfl(w, k, 32);
    int   mk = __shfl(m, k, 32);
    const float* xr = Xf + ((size_t)b * N_ + mk) * C_;
#pragma unroll
    for (int j = 0; j < 8; ++j) acc[j] += wk * xr[lane * 8 + j];
  }
  PK8 pk;
#pragma unroll
  for (int j = 0; j < 8; ++j) pk.h[j] = (bf16_t)acc[j];
  *reinterpret_cast<uint4*>(U2 + (size_t)wid * 512 + 256 + lane * 8) = pk.q;  // xj half
}

// ---- final BN + ReLU + residual, transpose back to NCHW -------------------
__global__ void k_out(const float* __restrict__ Hpre, const float* __restrict__ Xf,
                      const float* __restrict__ mean, const float* __restrict__ rstd,
                      const float* __restrict__ g, const float* __restrict__ bet,
                      float* __restrict__ out) {
  int i = blockIdx.x * blockDim.x + threadIdx.x;   // (b,o,p), p fastest
  if (i >= B_ * C_ * N_) return;
  int p = i & (N_ - 1); int o = (i >> 12) & 255; int b = i >> 20;
  size_t src = ((size_t)b * N_ + p) * C_ + o;
  float h = (Hpre[src] - mean[o]) * rstd[o] * g[o] + bet[o];
  h = fmaxf(h, 0.f);
  out[i] = h + Xf[src];
}

// ---------------------------------------------------------------------------
extern "C" void kernel_launch(void* const* d_in, const int* in_sizes, int n_in,
                              void* d_out, int out_size, void* d_ws, size_t ws_size,
                              hipStream_t stream) {
  const float* x1     = (const float*)d_in[0];
  const float* x2     = (const float*)d_in[1];
  const float* up_tw  = (const float*)d_in[2];
  const float* up_tb  = (const float*)d_in[3];
  const float* upc_w  = (const float*)d_in[4];
  const float* upc_b  = (const float*)d_in[5];
  const float* upc_g  = (const float*)d_in[6];
  const float* upc_be = (const float*)d_in[7];
  const float* emb_w  = (const float*)d_in[8];
  const float* emb_b  = (const float*)d_in[9];
  const float* att_w  = (const float*)d_in[10];
  const float* att_b  = (const float*)d_in[11];
  const float* ga_w   = (const float*)d_in[12];
  const float* ga_b   = (const float*)d_in[13];
  const float* ga_g   = (const float*)d_in[14];
  const float* ga_be  = (const float*)d_in[15];
  float* out = (float*)d_out;

  char* w = (char*)d_ws;
  size_t off = 0;
  auto alloc = [&](size_t bytes) -> void* {
    void* p = w + off;
    off = (off + bytes + 255) & ~(size_t)255;
    return p;
  };
  bf16_t* X1t  = (bf16_t*)alloc((size_t)B_ * N1_ * C_ * 2);
  bf16_t* U    = (bf16_t*)alloc((size_t)B_ * N_ * 512 * 2);   // [x2 | y]
  bf16_t* U2   = (bf16_t*)alloc((size_t)B_ * N_ * 512 * 2);   // [xn | xj]
  bf16_t* wt4  = (bf16_t*)alloc((size_t)4 * C_ * C_ * 2);
  bf16_t* wab  = (bf16_t*)alloc((size_t)C_ * 512 * 2);
  bf16_t* gaw  = (bf16_t*)alloc((size_t)C_ * 512 * 2);
  float*  Z    = (float*) alloc((size_t)B_ * N_ * C_ * 4);
  float*  Xf   = (float*) alloc((size_t)B_ * N_ * C_ * 4);
  float*  Hpre = (float*) alloc((size_t)B_ * N_ * C_ * 4);
  bf16_t* NA   = (bf16_t*)alloc((size_t)B_ * N_ * C_ * 2);
  int*    nnid = (int*)   alloc((size_t)B_ * N_ * KNN * 4);
  float*  mean1 = (float*)alloc(C_ * 4);
  float*  rstd1 = (float*)alloc(C_ * 4);
  float*  mean2 = (float*)alloc(C_ * 4);
  float*  rstd2 = (float*)alloc(C_ * 4);
  float*  invn  = (float*)alloc(B_ * C_ * 4);
  float*  va1   = (float*)alloc(C_ * 4);
  float*  va2   = (float*)alloc(C_ * 4);
  float*  c12   = (float*)alloc(2 * 4);
  float*  sqv   = (float*)alloc((size_t)B_ * N_ * 4);
  float*  a1    = (float*)alloc((size_t)B_ * N_ * 4);
  float*  a2    = (float*)alloc((size_t)B_ * N_ * 4);
  (void)ws_size; (void)in_sizes; (void)n_in; (void)out_size;

  const int TB = 256;
  // prep
  k_pack_x <<<(B_*C_*N1_ + B_*C_*N_ + TB - 1) / TB, TB, 0, stream>>>(x1, x2, X1t, U);
  k_pack_w <<<(4*C_*C_ + 2*C_*512 + TB - 1) / TB, TB, 0, stream>>>(up_tw, upc_w, ga_w,
                                                                   wt4, wab, gaw);
  k_attfold<<<1, C_, 0, stream>>>(emb_w, emb_b, att_w, va1, va2, c12);
  // ConvTranspose -> y half of U   (2*4*64*16 tiles / 8 waves per block)
  k_up<<<1024, TB, 0, stream>>>(X1t, wt4, up_tb, U);
  // conv1 (K=512) + BN + ReLU
  k_gemm_bias<<<1024, TB, 0, stream>>>(U, wab, upc_b, Z);
  k_colstats <<<C_, TB, 0, stream>>>(Z, mean1, rstd1);
  k_bnrelu   <<<(B_*N_*C_ + TB - 1) / TB, TB, 0, stream>>>(Z, mean1, rstd1, upc_g, upc_be,
                                                           Xf, U2);
  // normalization + folded attention scalars
  k_colnorm <<<B_*C_, TB, 0, stream>>>(Xf, invn);
  k_rowstats<<<1024, TB, 0, stream>>>(Xf, invn, va1, va2, c12, NA, sqv, a1, a2);
  // Gram + top-16 (fused, WMMA)
  k_knn<<<B_*256, 128, 0, stream>>>(NA, sqv, nnid);
  // attention + aggregation -> xj half of U2
  k_attn_agg<<<1024, TB, 0, stream>>>(Xf, nnid, a1, a2, att_b, U2);
  // final conv (K=512) + BN + ReLU + residual, NCHW out
  k_gemm_bias<<<1024, TB, 0, stream>>>(U2, gaw, ga_b, Hpre);
  k_colstats <<<C_, TB, 0, stream>>>(Hpre, mean2, rstd2);
  k_out<<<(B_*C_*N_ + TB - 1) / TB, TB, 0, stream>>>(Hpre, Xf, mean2, rstd2, ga_g, ga_be, out);
}
